// DeepSeekMoELayer_38328288149678
// MI455X (gfx1250) — matrix operand order repaired and checked by reading, not compiled, not run on hardware
//
#include <hip/hip_runtime.h>
#include <hip/hip_bf16.h>
#include <math.h>

typedef __bf16 bf16_t;
typedef __attribute__((ext_vector_type(16))) __bf16 v16bf;
typedef __attribute__((ext_vector_type(8)))  __bf16 v8bf;
typedef __attribute__((ext_vector_type(4)))  __bf16 v4bf;
typedef __attribute__((ext_vector_type(8)))  float  v8f;

#define E_EXP 64
#define TOPK  6
#define NSH   2
#define DIM   512
#define FDIM  512
#define NTOK  2048
#define NMAT  (3 * E_EXP + 3 * NSH)          /* 198 transposed weight matrices */
#define MATSZ ((size_t)DIM * FDIM)           /* elements per matrix            */

// ---------------------------------------------------------------- convert x -> bf16
__global__ __launch_bounds__(256) void moe_cvt_kernel(const float* __restrict__ x,
                                                      bf16_t* __restrict__ xb, int n) {
    int i = blockIdx.x * 256 + threadIdx.x;
    if (i < n) xb[i] = (bf16_t)x[i];
}

// ---------------------------------------------------------------- weights: fp32 [K][N] -> bf16 [N][K]
// mat id: [0,64) Wg | [64,128) Wu | [128,192) Wd | 192-3 sWg | 194-5 sWu | 196-7 sWd
__global__ __launch_bounds__(256) void moe_wt_kernel(
    const float* __restrict__ Wg, const float* __restrict__ Wu, const float* __restrict__ Wd,
    const float* __restrict__ sWg, const float* __restrict__ sWu, const float* __restrict__ sWd,
    bf16_t* __restrict__ WT) {
    int mat  = blockIdx.x;
    int tile = blockIdx.y;                       // 8x8 grid of 64x64 tiles
    const float* src;
    if      (mat < 64)  src = Wg  + (size_t)mat * MATSZ;
    else if (mat < 128) src = Wu  + (size_t)(mat - 64) * MATSZ;
    else if (mat < 192) src = Wd  + (size_t)(mat - 128) * MATSZ;
    else if (mat < 194) src = sWg + (size_t)(mat - 192) * MATSZ;
    else if (mat < 196) src = sWu + (size_t)(mat - 194) * MATSZ;
    else                src = sWd + (size_t)(mat - 196) * MATSZ;
    bf16_t* dst = WT + (size_t)mat * MATSZ;

    __shared__ float T[64][65];
    const int tk0 = (tile >> 3) * 64, tn0 = (tile & 7) * 64;
    const int tid = threadIdx.x;
    for (int i = tid; i < 64 * 16; i += 256) {            // coalesced float4 in
        int r = i >> 4, c4 = (i & 15) << 2;
        float4 v = *(const float4*)&src[(size_t)(tk0 + r) * 512 + tn0 + c4];
        T[r][c4 + 0] = v.x; T[r][c4 + 1] = v.y; T[r][c4 + 2] = v.z; T[r][c4 + 3] = v.w;
    }
    __syncthreads();
    for (int i = tid; i < 64 * 16; i += 256) {            // coalesced b64 bf16 out
        int n = i >> 4, k4 = (i & 15) << 2;
        v4bf h;
        h[0] = (bf16_t)T[k4 + 0][n]; h[1] = (bf16_t)T[k4 + 1][n];
        h[2] = (bf16_t)T[k4 + 2][n]; h[3] = (bf16_t)T[k4 + 3][n];
        *(v4bf*)&dst[(size_t)(tn0 + n) * 512 + tk0 + k4] = h;
    }
}

// ---------------------------------------------------------------- router + top-k
__global__ __launch_bounds__(64) void moe_router_kernel(const float* __restrict__ x,
                                                        const float* __restrict__ Wr,
                                                        float* __restrict__ topk_w,
                                                        int* __restrict__ counts,
                                                        int* __restrict__ lists) {
    int t = blockIdx.x;
    int e = threadIdx.x;
    __shared__ float xrow[DIM];
    __shared__ float logit[E_EXP];
    for (int d = e; d < DIM; d += 64) xrow[d] = x[(size_t)t * DIM + d];
    __syncthreads();
    float acc = 0.f;
    for (int d = 0; d < DIM; ++d) acc += xrow[d] * Wr[(size_t)d * E_EXP + e];
    logit[e] = acc;
    __syncthreads();
    if (e == 0) {
        float sv[TOPK]; int si[TOPK];
        for (int k = 0; k < TOPK; ++k) {
            float bv = -3.4e38f; int bi = 0;
            for (int j = 0; j < E_EXP; ++j)
                if (logit[j] > bv) { bv = logit[j]; bi = j; }
            sv[k] = bv; si[k] = bi; logit[bi] = -3.4e38f;
        }
        float m = sv[0], s = 0.f, w[TOPK];
        for (int k = 0; k < TOPK; ++k) { w[k] = __expf(sv[k] - m); s += w[k]; }
        float inv = 1.f / s;
        for (int k = 0; k < TOPK; ++k) {
            topk_w[t * TOPK + k] = w[k] * inv;
            int pos = atomicAdd(&counts[si[k]], 1);
            lists[(size_t)si[k] * NTOK + pos] = t * 8 + k;    // token*8 + k-slot
        }
    }
}

// ---------------------------------------------------------------- fragment helpers
__device__ __forceinline__ v16bf load_a_frag(const bf16_t* row, int kb, int lg) {
    // A 16x32 bf16: lane's 16 halves = K kb+lg*16 .. +15, contiguous
    v8bf a0 = *(const v8bf*)(row + kb + lg * 16);
    v8bf a1 = *(const v8bf*)(row + kb + lg * 16 + 8);
    return __builtin_shufflevector(a0, a1, 0, 1, 2, 3, 4, 5, 6, 7,
                                   8, 9, 10, 11, 12, 13, 14, 15);
}

// ---------------------------------------------------------------- per-expert SwiGLU FFN (WMMA bf16)
__global__ __launch_bounds__(256) void moe_ffn_kernel(
    const bf16_t* __restrict__ xb, const bf16_t* __restrict__ WT,
    const int* __restrict__ counts, const int* __restrict__ lists,
    float* __restrict__ routed_buf, float* __restrict__ shared_buf) {

    const int slot = blockIdx.x;              // 0..E_EXP+NSH-1
    const int tile = blockIdx.y;              // 0..127
    const bool isShared = (slot >= E_EXP);
    const int cnt  = isShared ? NTOK : counts[slot];
    const int row0 = tile * 16;
    if (row0 >= cnt) return;

    const int ss = slot - E_EXP;
    const bf16_t* wtg = WT + (size_t)(isShared ? 192 + ss : slot) * MATSZ;       // [F][D]
    const bf16_t* wtu = WT + (size_t)(isShared ? 194 + ss : 64 + slot) * MATSZ;  // [F][D]
    const bf16_t* wtd = WT + (size_t)(isShared ? 196 + ss : 128 + slot) * MATSZ; // [D][F]

    __shared__ __align__(32) bf16_t Xs[16 * DIM];    // 16 KB, [m][k]
    __shared__ __align__(32) bf16_t Hs[16 * FDIM];   // 16 KB, [m][f]
    __shared__ int rowTok[16];

    const int tid = threadIdx.x;
    if (tid < 16) {
        int r = tid, entry = -1;
        if (row0 + r < cnt)
            entry = isShared ? ((row0 + r) * 8) : lists[(size_t)slot * NTOK + row0 + r];
        rowTok[r] = entry;
    }
    __syncthreads();

    for (int i = tid; i < 16 * DIM; i += 256) {
        int r = i >> 9, d = i & (DIM - 1);
        int entry = rowTok[r];
        Xs[i] = (entry >= 0) ? xb[(size_t)(entry >> 3) * DIM + d] : (bf16_t)0.0f;
    }
    __syncthreads();

    const int wv   = tid >> 5;                // wave 0..7 owns 64 output columns
    const int lane = tid & 31;
    const int ln   = lane & 15;
    const int lg   = lane >> 4;
    const int fbase = wv * 64;

    // ---------------- gate + up ----------------
    v8f accg[4] = {}; v8f accu[4] = {};
    for (int kb = 0; kb < DIM; kb += 32) {
        v16bf a = load_a_frag(&Xs[ln * DIM], kb, lg);
#pragma unroll
        for (int s = 0; s < 4; ++s) {
            int col = fbase + s * 16 + ln;
            // B 32x16 bf16: lane's 16 halves = K kb+lg*16 .. +15, contiguous
            v16bf bg = *(const v16bf*)(wtg + (size_t)col * DIM + kb + lg * 16);
            v16bf bu = *(const v16bf*)(wtu + (size_t)col * DIM + kb + lg * 16);
            accg[s] = __builtin_amdgcn_wmma_f32_16x16x32_bf16(false, a, false, bg,
                                                              (short)0, accg[s], false, false);
            accu[s] = __builtin_amdgcn_wmma_f32_16x16x32_bf16(false, a, false, bu,
                                                              (short)0, accu[s], false, false);
        }
    }

    // H = silu(g) * u -> LDS (bf16)
#pragma unroll
    for (int s = 0; s < 4; ++s) {
        int col = fbase + s * 16 + ln;
#pragma unroll
        for (int v = 0; v < 8; ++v) {
            int m = v + lg * 8;
            float g = accg[s][v], u = accu[s][v];
            float h = (g / (1.f + __expf(-g))) * u;
            Hs[m * FDIM + col] = (bf16_t)h;
        }
    }
    __syncthreads();

    // ---------------- down projection ----------------
    v8f accd[4] = {};
    for (int kb = 0; kb < FDIM; kb += 32) {
        v16bf a = load_a_frag(&Hs[ln * FDIM], kb, lg);
#pragma unroll
        for (int s = 0; s < 4; ++s) {
            int col = fbase + s * 16 + ln;
            v16bf b = *(const v16bf*)(wtd + (size_t)col * FDIM + kb + lg * 16);
            accd[s] = __builtin_amdgcn_wmma_f32_16x16x32_bf16(false, a, false, b,
                                                              (short)0, accd[s], false, false);
        }
    }

    // scatter: each (token,k) slot written by exactly one block/lane -> deterministic
#pragma unroll
    for (int s = 0; s < 4; ++s) {
        int col = fbase + s * 16 + ln;
#pragma unroll
        for (int v = 0; v < 8; ++v) {
            int m = v + lg * 8;
            int entry = rowTok[m];
            if (entry < 0) continue;
            int t = entry >> 3, k = entry & 7;
            float val = accd[s][v];
            if (isShared)
                shared_buf[((size_t)ss * NTOK + t) * DIM + col] = val;
            else
                routed_buf[((size_t)t * TOPK + k) * DIM + col] = val;
        }
    }
}

// ---------------------------------------------------------------- combine
__global__ __launch_bounds__(256) void moe_combine_kernel(const float* __restrict__ x,
                                                          const float* __restrict__ topk_w,
                                                          const float* __restrict__ routed_buf,
                                                          const float* __restrict__ shared_buf,
                                                          float* __restrict__ out) {
    int t = blockIdx.x;
    float w[TOPK];
#pragma unroll
    for (int k = 0; k < TOPK; ++k) w[k] = topk_w[t * TOPK + k];
    for (int d = threadIdx.x; d < DIM; d += 256) {
        float acc = x[(size_t)t * DIM + d];
        acc += shared_buf[(size_t)t * DIM + d];
        acc += shared_buf[((size_t)NTOK + t) * DIM + d];
#pragma unroll
        for (int k = 0; k < TOPK; ++k)
            acc += w[k] * routed_buf[((size_t)t * TOPK + k) * DIM + d];
        out[(size_t)t * DIM + d] = acc;
    }
}

// ---------------------------------------------------------------- launch
extern "C" void kernel_launch(void* const* d_in, const int* in_sizes, int n_in,
                              void* d_out, int out_size, void* d_ws, size_t ws_size,
                              hipStream_t stream) {
    const float* x   = (const float*)d_in[0];
    const float* Wr  = (const float*)d_in[1];
    const float* sWg = (const float*)d_in[2];
    const float* sWu = (const float*)d_in[3];
    const float* sWd = (const float*)d_in[4];
    const float* Wg  = (const float*)d_in[5];
    const float* Wu  = (const float*)d_in[6];
    const float* Wd  = (const float*)d_in[7];
    float* out = (float*)d_out;

    char* ws = (char*)d_ws;
    size_t off = 0;
    auto take = [&](size_t bytes) -> char* {
        char* p = ws + off;
        off = (off + bytes + 255) & ~(size_t)255;
        return p;
    };
    bf16_t* WT      = (bf16_t*)take((size_t)NMAT * MATSZ * sizeof(bf16_t));  // ~104 MB
    bf16_t* xb      = (bf16_t*)take((size_t)NTOK * DIM * sizeof(bf16_t));
    int*    counts  = (int*)   take((size_t)E_EXP * sizeof(int));
    int*    lists   = (int*)   take((size_t)E_EXP * NTOK * sizeof(int));
    float*  topkw   = (float*) take((size_t)NTOK * TOPK * sizeof(float));
    float*  routedb = (float*) take((size_t)NTOK * TOPK * DIM * sizeof(float));
    float*  sharedb = (float*) take((size_t)NSH * NTOK * DIM * sizeof(float));

    hipMemsetAsync(counts, 0, E_EXP * sizeof(int), stream);

    int ncvt = NTOK * DIM;
    moe_cvt_kernel<<<(ncvt + 255) / 256, 256, 0, stream>>>(x, xb, ncvt);
    dim3 gwt(NMAT, 64);
    moe_wt_kernel<<<gwt, 256, 0, stream>>>(Wg, Wu, Wd, sWg, sWu, sWd, WT);
    moe_router_kernel<<<NTOK, 64, 0, stream>>>(x, Wr, topkw, counts, lists);
    dim3 gffn(E_EXP + NSH, NTOK / 16);
    moe_ffn_kernel<<<gffn, 256, 0, stream>>>(xb, WT, counts, lists, routedb, sharedb);
    moe_combine_kernel<<<NTOK, 256, 0, stream>>>(x, topkw, routedb, sharedb, out);
}
